// CrossAttention_76647986364914
// MI455X (gfx1250) — compile-verified, matrix operand order checked by reading, not run on hardware
//
#include <hip/hip_runtime.h>

typedef __attribute__((ext_vector_type(16))) _Float16 v16h;
typedef __attribute__((ext_vector_type(8)))  _Float16 v8h;
typedef __attribute__((ext_vector_type(8)))  float    v8f;
typedef __attribute__((ext_vector_type(4)))  float    f4;

__device__ __forceinline__ v8f wmma_f16(v16h a, v16h b, v8f c) {
  // D(f32 16x16) = A(f16 16x32) * B(f16 32x16) + C
  return __builtin_amdgcn_wmma_f32_16x16x32_f16(false, a, false, b, (short)0, c,
                                                false, false);
}
__device__ __forceinline__ v16h cat8(v8h lo, v8h hi) {
  return __builtin_shufflevector(lo, hi, 0, 1, 2, 3, 4, 5, 6, 7,
                                 8, 9, 10, 11, 12, 13, 14, 15);
}
// workgroup-relative LDS byte offset (low 32 bits of the flat address)
__device__ __forceinline__ unsigned lds_off(const void* p) {
  return (unsigned)(size_t)p;
}

// ---------------------------------------------------------------------------
// Weight pre-pass: Wt(f16, 512 x K, n-major) = transpose(W(f32, K x 512))
// ---------------------------------------------------------------------------
__global__ __launch_bounds__(256) void cvt_wt(const float* __restrict__ W,
                                              _Float16* __restrict__ Wt,
                                              int kshift) {
  int o = blockIdx.x * 256 + threadIdx.x;     // o = n*K + k
  int K = 1 << kshift;
  int n = o >> kshift, k = o & (K - 1);
  Wt[o] = (_Float16)W[((size_t)k << 9) + n];
}

// ---------------------------------------------------------------------------
// Projection GEMM: Out(f16, M x 512) = A(f32, batch-strided rows x K) * W
// W supplied as transposed f16 (512 x K). Each wave: 16 rows x 128 cols.
// B tiles staged in LDS via CDNA5 async global->LDS copies.
// store_mode 0: Out[row*512 + col]
// store_mode 1: transposed V store: Out[((b*8+h)*64+d)*96 + jbase + local_row]
// ---------------------------------------------------------------------------
__global__ __launch_bounds__(256) void proj_gemm(
    const float* __restrict__ A, const _Float16* __restrict__ Wt,
    _Float16* __restrict__ Out,
    int rpb, int bstride, int row_off, int K, int Mtiles,
    int store_mode, int jbase)
{
  __shared__ __align__(16) _Float16 wl[128 * 40];   // 128 cols x 32 k, pad 8
  const int tid = threadIdx.x;
  const int wave = tid >> 5, lane = tid & 31;
  const int m = lane & 15, g = lane >> 4;
  const int row_tile = blockIdx.x * 8 + wave;
  const bool active = row_tile < Mtiles;
  const int col0 = blockIdx.y * 128;

  const int gr = active ? row_tile * 16 + m : m;
  const int bb = gr / rpb;
  const int loc = gr - bb * rpb;
  const float* arow = A + ((size_t)bb * bstride + row_off + loc) * (size_t)K;

  // async-copy assignments: 512 16B chunks (128 rows x 64B), 2 per thread
  const int cn = tid >> 2;                 // chunk row 0..63
  const int ck = (tid & 3) * 8;            // k offset (f16)
  const unsigned ldsa0 = lds_off(&wl[cn * 40 + ck]);
  const unsigned ldsa1 = lds_off(&wl[(cn + 64) * 40 + ck]);
  const unsigned gb0 = (unsigned)(((col0 + cn) * K + ck) * 2);
  const unsigned gb1 = (unsigned)(((col0 + cn + 64) * K + ck) * 2);

  v8f acc[8] = {};
  for (int k0 = 0; k0 < K; k0 += 32) {
    __syncthreads();
    unsigned vo0 = gb0 + (unsigned)(k0 * 2);
    unsigned vo1 = gb1 + (unsigned)(k0 * 2);
    asm volatile("global_load_async_to_lds_b128 %0, %1, %2"
                 :: "v"(ldsa0), "v"(vo0), "s"(Wt) : "memory");
    asm volatile("global_load_async_to_lds_b128 %0, %1, %2"
                 :: "v"(ldsa1), "v"(vo1), "s"(Wt) : "memory");
    asm volatile("s_wait_asynccnt 0" ::: "memory");
    __syncthreads();

    const float* ap = arow + k0 + g * 8;
    f4 a0 = *(const f4*)ap;
    f4 a1 = *(const f4*)(ap + 4);
    f4 a2 = *(const f4*)(ap + 16);
    f4 a3 = *(const f4*)(ap + 20);
    v16h a;
#pragma unroll
    for (int j = 0; j < 4; ++j) {
      a[j]      = (_Float16)a0[j];
      a[4 + j]  = (_Float16)a1[j];
      a[8 + j]  = (_Float16)a2[j];
      a[12 + j] = (_Float16)a3[j];
    }
#pragma unroll
    for (int t = 0; t < 8; ++t) {
      const _Float16* wr = &wl[(t * 16 + m) * 40];
      v16h bf = cat8(*(const v8h*)(wr + g * 8), *(const v8h*)(wr + 16 + g * 8));
      acc[t] = wmma_f16(a, bf, acc[t]);
    }
  }
  if (!active) return;

#pragma unroll
  for (int t = 0; t < 8; ++t) {
#pragma unroll
    for (int r = 0; r < 8; ++r) {
      int orow = row_tile * 16 + r + 8 * g;   // C layout: M = r + 8*g, N = m
      int ocol = col0 + t * 16 + m;
      if (store_mode == 0) {
        Out[(size_t)orow * 512 + ocol] = (_Float16)acc[t][r];
      } else {
        int b2 = orow / rpb, loc2 = orow - b2 * rpb;
        int hh = ocol >> 6, dd = ocol & 63;
        Out[((size_t)(b2 * 8 + hh) * 64 + dd) * 96 + jbase + loc2] =
            (_Float16)acc[t][r];
      }
    }
  }
}

// ---------------------------------------------------------------------------
// Fused two-source attention: per (b, h, 16-row q tile)
// scores(txt 16x80 masked @77, ip 16x16) -> 2 softmaxes -> P_txt*V + P_ip*V_ip
// vt layout: [((b*8+h)*64 + d)*96 + j], j in [0,77) txt, [80,96) ip
// ---------------------------------------------------------------------------
__global__ __launch_bounds__(128) void attn_kernel(
    const _Float16* __restrict__ q, const _Float16* __restrict__ k,
    const _Float16* __restrict__ kip, const _Float16* __restrict__ vt,
    _Float16* __restrict__ o)
{
  __shared__ __align__(16) _Float16 lp[4][16 * 104];
  const int wave = threadIdx.x >> 5, lane = threadIdx.x & 31;
  const int m = lane & 15, g = lane >> 4;
  const int b = blockIdx.z, h = blockIdx.y;
  const int qtile = blockIdx.x * 4 + wave;  // 0..159

  // q A-fragments (K=64 -> two 16x32 frags), pre-scaled by 1/sqrt(64)
  const _Float16* qrow =
      q + ((size_t)(b * 2560 + qtile * 16 + m)) * 512 + h * 64;
  v16h aq0 = cat8(*(const v8h*)(qrow + g * 8), *(const v8h*)(qrow + 16 + g * 8));
  v16h aq1 = cat8(*(const v8h*)(qrow + 32 + g * 8), *(const v8h*)(qrow + 48 + g * 8));
  aq0 = aq0 * (_Float16)0.125f;
  aq1 = aq1 * (_Float16)0.125f;

  // ---- scores: text (5 tiles of 16 keys, mask j>=77) + ip (1 tile) ----
  v8f s[5], sip;
#pragma unroll
  for (int jt = 0; jt < 5; ++jt) {
    v16h b0 = {}, b1 = {};
    int j = jt * 16 + m;                    // this lane's key column
    if (j < 77) {
      const _Float16* kr = k + ((size_t)(b * 77 + j)) * 512 + h * 64;
      b0 = cat8(*(const v8h*)(kr + g * 8), *(const v8h*)(kr + 16 + g * 8));
      b1 = cat8(*(const v8h*)(kr + 32 + g * 8), *(const v8h*)(kr + 48 + g * 8));
    }
    v8f z = {};
    z = wmma_f16(aq0, b0, z);
    s[jt] = wmma_f16(aq1, b1, z);
  }
  if (m >= 13) {                            // columns 77..79 -> -inf
#pragma unroll
    for (int r = 0; r < 8; ++r) s[4][r] = -1e30f;
  }
  {
    const _Float16* kr = kip + ((size_t)(b * 16 + m)) * 512 + h * 64;
    v16h b0 = cat8(*(const v8h*)(kr + g * 8), *(const v8h*)(kr + 16 + g * 8));
    v16h b1 = cat8(*(const v8h*)(kr + 32 + g * 8), *(const v8h*)(kr + 48 + g * 8));
    v8f z = {};
    z = wmma_f16(aq0, b0, z);
    sip = wmma_f16(aq1, b1, z);
  }

  // ---- two independent softmaxes, normalized in place ----
#pragma unroll
  for (int r = 0; r < 8; ++r) {
    float mx = -1e30f;
#pragma unroll
    for (int jt = 0; jt < 5; ++jt) mx = fmaxf(mx, s[jt][r]);
#pragma unroll
    for (int d = 1; d < 16; d <<= 1) mx = fmaxf(mx, __shfl_xor(mx, d, 32));
    float sum = 0.f;
#pragma unroll
    for (int jt = 0; jt < 5; ++jt) {
      float p = __expf(s[jt][r] - mx);
      s[jt][r] = p;
      sum += p;
    }
#pragma unroll
    for (int d = 1; d < 16; d <<= 1) sum += __shfl_xor(sum, d, 32);
    float inv = 1.0f / sum;
#pragma unroll
    for (int jt = 0; jt < 5; ++jt) s[jt][r] *= inv;

    float mxi = sip[r];
#pragma unroll
    for (int d = 1; d < 16; d <<= 1) mxi = fmaxf(mxi, __shfl_xor(mxi, d, 32));
    float pi = __expf(sip[r] - mxi);
    float sumi = pi;
#pragma unroll
    for (int d = 1; d < 16; d <<= 1) sumi += __shfl_xor(sumi, d, 32);
    sip[r] = pi / sumi;
  }

  // ---- transpose P (C layout) -> A-frag layout via LDS ----
  _Float16* myp = lp[wave];
#pragma unroll
  for (int jt = 0; jt < 5; ++jt)
#pragma unroll
    for (int r = 0; r < 8; ++r)
      myp[(r + 8 * g) * 104 + jt * 16 + m] = (_Float16)s[jt][r];
#pragma unroll
  for (int r = 0; r < 8; ++r)
    myp[(r + 8 * g) * 104 + 80 + m] = (_Float16)sip[r];
  asm volatile("s_wait_dscnt 0" ::: "memory");

  // ---- O = P * V  (K = 96: txt 0..79, ip 80..95), fused accumulation ----
  v8f oc[4] = {};
#pragma unroll
  for (int ks = 0; ks < 3; ++ks) {
    const _Float16* pr = myp + m * 104 + ks * 32;
    v16h pa = cat8(*(const v8h*)(pr + g * 8), *(const v8h*)(pr + 16 + g * 8));
#pragma unroll
    for (int t = 0; t < 4; ++t) {
      const _Float16* vr =
          vt + ((size_t)(b * 8 + h) * 64 + t * 16 + m) * 96 + ks * 32;
      v16h bv = cat8(*(const v8h*)(vr + g * 8), *(const v8h*)(vr + 16 + g * 8));
      oc[t] = wmma_f16(pa, bv, oc[t]);
    }
  }

#pragma unroll
  for (int t = 0; t < 4; ++t)
#pragma unroll
    for (int r = 0; r < 8; ++r)
      o[((size_t)(b * 2560 + qtile * 16 + r + 8 * g)) * 512 + h * 64 + t * 16 + m] =
          (_Float16)oc[t][r];
}

// ---------------------------------------------------------------------------
// Output GEMM: out(f32, 81920 x 512) = attn(f16) * W_out^T(f16 512x512) + b
// ---------------------------------------------------------------------------
__global__ __launch_bounds__(256) void out_gemm(
    const _Float16* __restrict__ A, const _Float16* __restrict__ Wt,
    const float* __restrict__ bias, float* __restrict__ Out, int Mtiles)
{
  __shared__ __align__(16) _Float16 wl[128 * 40];
  const int tid = threadIdx.x;
  const int wave = tid >> 5, lane = tid & 31;
  const int m = lane & 15, g = lane >> 4;
  const int row_tile = blockIdx.x * 8 + wave;
  const bool active = row_tile < Mtiles;
  const int col0 = blockIdx.y * 128;
  const int gr = active ? row_tile * 16 + m : m;
  const _Float16* arow = A + (size_t)gr * 512;

  const int cn = tid >> 2;
  const int ck = (tid & 3) * 8;
  const unsigned ldsa0 = lds_off(&wl[cn * 40 + ck]);
  const unsigned ldsa1 = lds_off(&wl[(cn + 64) * 40 + ck]);
  const unsigned gb0 = (unsigned)(((col0 + cn) * 512 + ck) * 2);
  const unsigned gb1 = (unsigned)(((col0 + cn + 64) * 512 + ck) * 2);

  v8f acc[8] = {};
  for (int k0 = 0; k0 < 512; k0 += 32) {
    __syncthreads();
    unsigned vo0 = gb0 + (unsigned)(k0 * 2);
    unsigned vo1 = gb1 + (unsigned)(k0 * 2);
    asm volatile("global_load_async_to_lds_b128 %0, %1, %2"
                 :: "v"(ldsa0), "v"(vo0), "s"(Wt) : "memory");
    asm volatile("global_load_async_to_lds_b128 %0, %1, %2"
                 :: "v"(ldsa1), "v"(vo1), "s"(Wt) : "memory");
    asm volatile("s_wait_asynccnt 0" ::: "memory");
    __syncthreads();

    const _Float16* ap = arow + k0 + g * 8;
    v16h a = cat8(*(const v8h*)ap, *(const v8h*)(ap + 16));
#pragma unroll
    for (int t = 0; t < 8; ++t) {
      const _Float16* wr = &wl[(t * 16 + m) * 40];
      v16h bf = cat8(*(const v8h*)(wr + g * 8), *(const v8h*)(wr + 16 + g * 8));
      acc[t] = wmma_f16(a, bf, acc[t]);
    }
  }
  if (!active) return;

#pragma unroll
  for (int t = 0; t < 8; ++t) {
    float bi = bias[col0 + t * 16 + m];
#pragma unroll
    for (int r = 0; r < 8; ++r) {
      int orow = row_tile * 16 + r + 8 * g;
      Out[(size_t)orow * 512 + col0 + t * 16 + m] = acc[t][r] + bi;
    }
  }
}

// ---------------------------------------------------------------------------
extern "C" void kernel_launch(void* const* d_in, const int* in_sizes, int n_in,
                              void* d_out, int out_size, void* d_ws, size_t ws_size,
                              hipStream_t stream) {
  const float* x    = (const float*)d_in[0];   // (32, 2560, 512)
  const float* ctx  = (const float*)d_in[1];   // (32, 93, 1024)
  const float* Wq   = (const float*)d_in[2];   // (512, 512)
  const float* Wk   = (const float*)d_in[3];   // (1024, 512)
  const float* Wv   = (const float*)d_in[4];
  const float* Wkip = (const float*)d_in[5];
  const float* Wvip = (const float*)d_in[6];
  const float* Wout = (const float*)d_in[7];   // (512, 512)
  const float* bout = (const float*)d_in[8];   // (512,)
  float* out = (float*)d_out;                  // (32, 2560, 512) f32

  char* ws = (char*)d_ws;
  size_t off = 0;
  auto alloc = [&](size_t bytes) {
    void* p = ws + off;
    off += (bytes + 255) & ~(size_t)255;
    return p;
  };
  _Float16* q_ws    = (_Float16*)alloc(81920ull * 512 * 2);        // 83.9 MB
  _Float16* k_ws    = (_Float16*)alloc(2464ull * 512 * 2);         //  2.5 MB
  _Float16* kip_ws  = (_Float16*)alloc(512ull * 512 * 2);          //  0.5 MB
  _Float16* vt_ws   = (_Float16*)alloc(32ull * 8 * 64 * 96 * 2);   //  3.1 MB
  _Float16* at_ws   = (_Float16*)alloc(81920ull * 512 * 2);        // 83.9 MB
  _Float16* wq_t    = (_Float16*)alloc(512ull * 512 * 2);
  _Float16* wk_t    = (_Float16*)alloc(512ull * 1024 * 2);
  _Float16* wv_t    = (_Float16*)alloc(512ull * 1024 * 2);
  _Float16* wkip_t  = (_Float16*)alloc(512ull * 1024 * 2);
  _Float16* wvip_t  = (_Float16*)alloc(512ull * 1024 * 2);
  _Float16* wout_t  = (_Float16*)alloc(512ull * 512 * 2);

  // zero transposed-V buffer so padding slots (j=77..79) multiply to 0 safely
  hipMemsetAsync(vt_ws, 0, 32ull * 8 * 64 * 96 * 2, stream);

  // weight transpose+convert passes (tiny)
  cvt_wt<<<1024, 256, 0, stream>>>(Wq,   wq_t,   9);
  cvt_wt<<<2048, 256, 0, stream>>>(Wk,   wk_t,   10);
  cvt_wt<<<2048, 256, 0, stream>>>(Wv,   wv_t,   10);
  cvt_wt<<<2048, 256, 0, stream>>>(Wkip, wkip_t, 10);
  cvt_wt<<<2048, 256, 0, stream>>>(Wvip, wvip_t, 10);
  cvt_wt<<<1024, 256, 0, stream>>>(Wout, wout_t, 9);

  dim3 blk(256);
  // q = x @ W_q            M=81920 (rpb 2560), K=512
  proj_gemm<<<dim3(640, 4), blk, 0, stream>>>(x, wq_t, q_ws, 2560, 2560, 0, 512, 5120, 0, 0);
  // k = ctx[:, :77] @ W_k  M=2464 (rpb 77, bstride 93), K=1024
  proj_gemm<<<dim3(20, 4), blk, 0, stream>>>(ctx, wk_t, k_ws, 77, 93, 0, 1024, 154, 0, 0);
  // v (transposed store into vt[..., 0:77])
  proj_gemm<<<dim3(20, 4), blk, 0, stream>>>(ctx, wv_t, vt_ws, 77, 93, 0, 1024, 154, 1, 0);
  // k_ip = ctx[:, 77:] @ W_k_ip  M=512 (rpb 16, offset 77)
  proj_gemm<<<dim3(4, 4), blk, 0, stream>>>(ctx, wkip_t, kip_ws, 16, 93, 77, 1024, 32, 0, 0);
  // v_ip (transposed store into vt[..., 80:96])
  proj_gemm<<<dim3(4, 4), blk, 0, stream>>>(ctx, wvip_t, vt_ws, 16, 93, 77, 1024, 32, 1, 80);
  // fused txt+ip attention
  attn_kernel<<<dim3(40, 8, 32), dim3(128), 0, stream>>>(q_ws, k_ws, kip_ws, vt_ws, at_ws);
  // out = attn @ W_out + b_out
  out_gemm<<<dim3(640, 4), blk, 0, stream>>>(at_ws, wout_t, bout, out, 5120);
}
